// Pointer_Decoder_56916906606845
// MI455X (gfx1250) — matrix operand order repaired and checked by reading, not compiled, not run on hardware
//
#include <hip/hip_runtime.h>
#include <hip/hip_bf16.h>
#include <math.h>

// Problem constants (from reference)
#define Bb   32
#define TD_  32
#define TK_  400
#define H_   512
#define E_   512
#define V_   50000
#define OOV_ 50
#define VX_  (V_ + OOV_)
#define H2_  1024          // 2H
#define KX_  1536          // 2H + E  (also H + 2H for [h|c_t])
#define G4_  2048          // 4H
#define PGK_ 2560          // E + 2H + 2H

typedef __bf16 bf16;
typedef __attribute__((ext_vector_type(16))) __bf16 v16bf;
typedef __attribute__((ext_vector_type(8)))  __bf16 v8bf;
typedef __attribute__((ext_vector_type(8)))  float  v8f;

union V16U { v16bf v; v8bf h[2]; };

// ---------------- warp / block reductions (wave32) ----------------
__device__ __forceinline__ float warpSum(float v) {
#pragma unroll
  for (int o = 16; o > 0; o >>= 1) v += __shfl_xor(v, o);
  return v;
}
__device__ __forceinline__ float warpMax(float v) {
#pragma unroll
  for (int o = 16; o > 0; o >>= 1) v = fmaxf(v, __shfl_xor(v, o));
  return v;
}
__device__ float blockSum(float v, float* red) {
  const int lane = threadIdx.x & 31, w = threadIdx.x >> 5, nw = blockDim.x >> 5;
  v = warpSum(v);
  if (lane == 0) red[w] = v;
  __syncthreads();
  float r = (lane < nw) ? red[lane] : 0.0f;
  if (w == 0) { r = warpSum(r); if (lane == 0) red[0] = r; }
  __syncthreads();
  float out = red[0];
  __syncthreads();
  return out;
}
__device__ float blockMax(float v, float* red) {
  const int lane = threadIdx.x & 31, w = threadIdx.x >> 5, nw = blockDim.x >> 5;
  v = warpMax(v);
  if (lane == 0) red[w] = v;
  __syncthreads();
  float r = (lane < nw) ? red[lane] : -1e30f;
  if (w == 0) { r = warpMax(r); if (lane == 0) red[0] = r; }
  __syncthreads();
  float out = red[0];
  __syncthreads();
  return out;
}

// ---------------- bf16 WMMA GEMM: D[m,n] = sum_k A[m,k]*W[n,k] (+bias[n]) ----
// A: bf16 [M,K] row-major, W: bf16 [N,K] row-major, D: f32 [M,ldD].
// Optional bf16 mirror Dbf with leading dim ldDbf.
// One wave per block computes a 32x16 tile; grid = (N/16, M/32).
__global__ __launch_bounds__(32)
void k_gemm_bf16(const bf16* __restrict__ A, const bf16* __restrict__ W,
                 const float* __restrict__ bias, float* __restrict__ D,
                 bf16* __restrict__ Dbf, int ldDbf, int K, int ldD)
{
  const int lane = threadIdx.x;
  const int half = lane >> 4;     // upper/lower half-wave
  const int l16  = lane & 15;
  const int n    = blockIdx.x * 16 + l16;
  const int m0   = blockIdx.y * 32;

  // ISA 16-bit A (16x32) layout: lanes<16 K chunks {k..k+7, k+16..k+23},
// lanes>=16 get +8 shifted chunks. B (32x16): lane = N, same K chunks as A,
// upper half-wave offset by 8.
  const bf16* wrow  = W + (size_t)n * K + half * 8;
  const bf16* arow0 = A + (size_t)(m0 + l16) * K + half * 8;
  const bf16* arow1 = arow0 + (size_t)16 * K;

  v8f acc0 = {}; v8f acc1 = {};
  for (int k = 0; k < K; k += 32) {
    V16U a0, a1, b;
    a0.h[0] = *(const v8bf*)(arow0 + k);
    a0.h[1] = *(const v8bf*)(arow0 + k + 16);
    a1.h[0] = *(const v8bf*)(arow1 + k);
    a1.h[1] = *(const v8bf*)(arow1 + k + 16);
    b.h[0]  = *(const v8bf*)(wrow + k);
    b.h[1]  = *(const v8bf*)(wrow + k + 16);
    __builtin_prefetch(wrow + k + 256, 0, 0);   // global_prefetch_b8 on weight stream
    acc0 = __builtin_amdgcn_wmma_f32_16x16x32_bf16(false, a0.v, false, b.v,
                                                   (short)0, acc0, false, false);
    acc1 = __builtin_amdgcn_wmma_f32_16x16x32_bf16(false, a1.v, false, b.v,
                                                   (short)0, acc1, false, false);
  }
  const float bv = bias ? bias[n] : 0.0f;
#pragma unroll
  for (int r = 0; r < 8; ++r) {     // C layout: VGPR r, M = m0 + r + 8*half
    const int ma = m0 + half * 8 + r;
    const float va = acc0[r] + bv;
    const float vb = acc1[r] + bv;
    D[(size_t)ma * ldD + n] = va;
    D[(size_t)(ma + 16) * ldD + n] = vb;
    if (Dbf) {
      Dbf[(size_t)ma * ldDbf + n] = (bf16)va;
      Dbf[(size_t)(ma + 16) * ldDbf + n] = (bf16)vb;
    }
  }
}

// ---------------- helpers: conversions / setup ----------------
__global__ void k_cvt(const float* __restrict__ s, bf16* __restrict__ d, int n) {
  int i = blockIdx.x * blockDim.x + threadIdx.x;
  const int stride = gridDim.x * blockDim.x;
  for (; i < n; i += stride) d[i] = (bf16)s[i];
}

__global__ void k_wcat(const float* __restrict__ wih, const float* __restrict__ whh,
                       bf16* __restrict__ wcat) {
  int i = blockIdx.x * blockDim.x + threadIdx.x;
  const int stride = gridDim.x * blockDim.x;
  for (; i < G4_ * H2_; i += stride) {
    const int g = i / H2_, k = i % H2_;
    wcat[i] = (bf16)(k < H_ ? wih[(size_t)g * H_ + k] : whh[(size_t)g * H_ + (k - H_)]);
  }
}

__global__ void k_bsum(const float* __restrict__ a, const float* __restrict__ b,
                       float* __restrict__ d) {
  const int i = blockIdx.x * blockDim.x + threadIdx.x;
  if (i < G4_) d[i] = a[i] + b[i];
}

__global__ void k_init(const float* __restrict__ h0, const float* __restrict__ c0,
                       const float* __restrict__ cov_in, float* __restrict__ h,
                       float* __restrict__ c, float* __restrict__ ctx,
                       float* __restrict__ cov, bf16* __restrict__ A2) {
  const int i = blockIdx.x * blockDim.x + threadIdx.x;
  if (i < Bb * H2_) ctx[i] = 0.0f;
  if (i < Bb * H_) {
    h[i] = h0[i]; c[i] = c0[i];
    const int b = i / H_, j = i % H_;
    A2[(size_t)b * H2_ + H_ + j] = (bf16)h0[i];
  }
  if (i < Bb * TK_) cov[i] = cov_in[i];
}

// xcat = bf16([context | embedding[inputs[:,step]]])  -> [B, KX]
__global__ void k_xcat(const float* __restrict__ ctx, const int* __restrict__ inputs,
                       const float* __restrict__ emb, bf16* __restrict__ xcat, int step) {
  const int i = blockIdx.x * blockDim.x + threadIdx.x;
  if (i >= Bb * KX_) return;
  const int b = i / KX_, j = i % KX_;
  float v;
  if (j < H2_) v = ctx[(size_t)b * H2_ + j];
  else         v = emb[(size_t)inputs[b * TD_ + step] * E_ + (j - H2_)];
  xcat[i] = (bf16)v;
}

// LSTM pointwise: gates [B,4H] (i,f,g,o). Updates h,c; fills bf16 staging.
__global__ void k_lstm(const float* __restrict__ gates, float* __restrict__ h,
                       float* __restrict__ c, bf16* __restrict__ A2,
                       bf16* __restrict__ Ash, bf16* __restrict__ Ahc,
                       float* __restrict__ h_out, float* __restrict__ c_out) {
  const int i = blockIdx.x * blockDim.x + threadIdx.x;
  if (i >= Bb * H_) return;
  const int b = i / H_, j = i % H_;
  const float* g = gates + (size_t)b * G4_;
  const float ig = 1.0f / (1.0f + expf(-g[j]));
  const float fg = 1.0f / (1.0f + expf(-g[H_ + j]));
  const float gg = tanhf(g[2 * H_ + j]);
  const float og = 1.0f / (1.0f + expf(-g[3 * H_ + j]));
  const float cn = fg * c[i] + ig * gg;
  const float hn = og * tanhf(cn);
  c[i] = cn; h[i] = hn;
  A2 [(size_t)b * H2_ + H_ + j] = (bf16)hn;   // [x|h] for gates GEMM (next step)
  Ash[(size_t)b * H2_ + j]       = (bf16)hn;  // state_hat = [h|c]
  Ash[(size_t)b * H2_ + H_ + j]  = (bf16)cn;
  Ahc[(size_t)b * KX_ + j]       = (bf16)hn;  // [h|c_t] for W1 GEMM
  h_out[i] = hn; c_out[i] = cn;               // final-state output slots
}

// Attention + coverage + context + p_gen. One block per batch element.
__global__ __launch_bounds__(256)
void k_attn(const float* __restrict__ encf, const float* __restrict__ decf,
            const float* __restrict__ Wc, const float* __restrict__ vw,
            const float* __restrict__ enc, const float* __restrict__ mask,
            const float* __restrict__ h, const float* __restrict__ c,
            const float* __restrict__ x, const float* __restrict__ Wpg,
            const float* __restrict__ bpg,
            float* __restrict__ cov, float* __restrict__ ctx,
            bf16* __restrict__ Ahc, float* __restrict__ attn_ws,
            float* __restrict__ attns_out, float* __restrict__ covs_out,
            float* __restrict__ pg_ws, float* __restrict__ pgens_out, int step)
{
  __shared__ float sc[TK_];
  __shared__ float cts[H2_];
  __shared__ float red[8];
  const int b = blockIdx.x, tid = threadIdx.x;
  const int lane = tid & 31, w = tid >> 5;

  // scores[t] = v . tanh(enc_feature + dec_fea + cov*Wc)
  for (int t = w; t < TK_; t += 8) {
    const float cv = cov[b * TK_ + t];
    const float* ef = encf + ((size_t)b * TK_ + t) * H2_;
    float part = 0.0f;
    for (int k = lane; k < H2_; k += 32)
      part += vw[k] * tanhf(ef[k] + decf[(size_t)b * H2_ + k] + cv * Wc[k]);
    part = warpSum(part);
    if (lane == 0) sc[t] = part;
  }
  __syncthreads();

  // masked softmax + renormalize (== softmax*mask/sum)
  float m = -1e30f;
  for (int t = tid; t < TK_; t += 256) m = fmaxf(m, sc[t]);
  m = blockMax(m, red);
  for (int t = tid; t < TK_; t += 256)
    sc[t] = expf(sc[t] - m) * mask[b * TK_ + t];
  float s = 0.0f;
  for (int t = tid; t < TK_; t += 256) s += sc[t];
  s = blockSum(s, red);
  const float inv = 1.0f / s;
  for (int t = tid; t < TK_; t += 256) {
    const float a = sc[t] * inv;
    sc[t] = a;
    const float cn = cov[b * TK_ + t] + a;
    cov[b * TK_ + t] = cn;
    attn_ws[b * TK_ + t] = a;
    attns_out[((size_t)b * TD_ + step) * TK_ + t] = a;
    covs_out [((size_t)b * TD_ + step) * TK_ + t] = cn;
  }
  __syncthreads();

  // c_t = attn @ encoder_outputs
  for (int n = tid; n < H2_; n += 256) {
    float acc = 0.0f;
    const float* e = enc + (size_t)b * TK_ * H2_ + n;
    for (int t = 0; t < TK_; ++t) acc += sc[t] * e[(size_t)t * H2_];
    cts[n] = acc;
    ctx[(size_t)b * H2_ + n] = acc;                 // next-step context
    Ahc[(size_t)b * KX_ + H_ + n] = (bf16)acc;      // [h|c_t] for W1
  }
  __syncthreads();

  // p_gen = sigmoid(Wpg . [c_t, h, c, x] + bpg)
  float p = 0.0f;
  for (int j = tid; j < PGK_; j += 256) {
    float val;
    if      (j < H2_)          val = cts[j];
    else if (j < H2_ + H_)     val = h[(size_t)b * H_ + (j - H2_)];
    else if (j < H2_ + 2 * H_) val = c[(size_t)b * H_ + (j - H2_ - H_)];
    else                       val = x[(size_t)b * H_ + (j - H2_ - 2 * H_)];
    p += val * Wpg[j];
  }
  p = blockSum(p, red);
  if (tid == 0) {
    const float g = 1.0f / (1.0f + expf(-(p + bpg[0])));
    pg_ws[b] = g;
    pgens_out[b * TD_ + step] = g;
  }
}

// Vocab softmax * p_gen, zero OOV tail, scatter-add (1-p_gen)*attn.
__global__ __launch_bounds__(256)
void k_vocab(const float* __restrict__ logits, const float* __restrict__ pg,
             const float* __restrict__ attn, const int* __restrict__ ebev,
             float* __restrict__ out, int step)
{
  __shared__ float red[8];
  const int b = blockIdx.x, tid = threadIdx.x;
  const float* lg = logits + (size_t)b * V_;
  float* orow = out + ((size_t)b * TD_ + step) * VX_;

  float m = -1e30f;
  for (int v = tid; v < V_; v += 256) m = fmaxf(m, lg[v]);
  m = blockMax(m, red);
  float s = 0.0f;
  for (int v = tid; v < V_; v += 256) s += expf(lg[v] - m);
  s = blockSum(s, red);
  const float p = pg[b];
  const float pscale = p / s;
  for (int v = tid; v < V_; v += 256) orow[v] = expf(lg[v] - m) * pscale;
  for (int v = V_ + tid; v < VX_; v += 256) orow[v] = 0.0f;
  __threadfence();
  __syncthreads();
  const float om = 1.0f - p;
  for (int t = tid; t < TK_; t += 256)
    atomicAdd(&orow[ebev[b * TK_ + t]], om * attn[b * TK_ + t]);
}

// ---------------- host launch ----------------
extern "C" void kernel_launch(void* const* d_in, const int* in_sizes, int n_in,
                              void* d_out, int out_size, void* d_ws, size_t ws_size,
                              hipStream_t stream)
{
  (void)in_sizes; (void)n_in; (void)out_size; (void)ws_size;
  const int*   inputs = (const int*)  d_in[0];
  const float* h0     = (const float*)d_in[1];
  const float* c0     = (const float*)d_in[2];
  const float* enc    = (const float*)d_in[3];
  const float* mask   = (const float*)d_in[4];
  /* d_in[5] extra_zeros: zeros, unused */
  const int*   ebev   = (const int*)  d_in[6];
  const float* cov_in = (const float*)d_in[7];
  const float* emb    = (const float*)d_in[8];
  const float* W_ih   = (const float*)d_in[9];
  const float* W_hh   = (const float*)d_in[10];
  const float* b_ih   = (const float*)d_in[11];
  const float* b_hh   = (const float*)d_in[12];
  const float* Wx     = (const float*)d_in[13];
  const float* bx     = (const float*)d_in[14];
  const float* Wpg    = (const float*)d_in[15];
  const float* bpg    = (const float*)d_in[16];
  const float* Wenc   = (const float*)d_in[17];
  const float* Wp     = (const float*)d_in[18];
  const float* bp     = (const float*)d_in[19];
  const float* v_w    = (const float*)d_in[20];
  const float* Wc     = (const float*)d_in[21];
  const float* W1     = (const float*)d_in[22];
  const float* b1     = (const float*)d_in[23];
  const float* W2     = (const float*)d_in[24];
  const float* b2     = (const float*)d_in[25];

  // workspace bump allocator
  size_t off = 0;
  auto alloc = [&](size_t bytes) -> void* {
    off = (off + 255) & ~(size_t)255;
    void* p = (char*)d_ws + off;
    off += bytes;
    return p;
  };
  bf16*  W2b    = (bf16*) alloc((size_t)V_ * H_ * 2);
  bf16*  Wxb    = (bf16*) alloc((size_t)E_ * KX_ * 2);
  bf16*  Wcatb  = (bf16*) alloc((size_t)G4_ * H2_ * 2);
  bf16*  Wpb    = (bf16*) alloc((size_t)H2_ * H2_ * 2);
  bf16*  W1b    = (bf16*) alloc((size_t)H_ * KX_ * 2);
  bf16*  Wencb  = (bf16*) alloc((size_t)H2_ * H2_ * 2);
  bf16*  encb   = (bf16*) alloc((size_t)Bb * TK_ * H2_ * 2);
  float* encf   = (float*)alloc((size_t)Bb * TK_ * H2_ * 4);
  float* bsum   = (float*)alloc((size_t)G4_ * 4);
  float* logits = (float*)alloc((size_t)Bb * V_ * 4);
  float* gates  = (float*)alloc((size_t)Bb * G4_ * 4);
  bf16*  xcat   = (bf16*) alloc((size_t)Bb * KX_ * 2);
  bf16*  A2     = (bf16*) alloc((size_t)Bb * H2_ * 2);
  bf16*  Ash    = (bf16*) alloc((size_t)Bb * H2_ * 2);
  bf16*  Ahc    = (bf16*) alloc((size_t)Bb * KX_ * 2);
  bf16*  out1b  = (bf16*) alloc((size_t)Bb * H_ * 2);
  float* xf     = (float*)alloc((size_t)Bb * H_ * 4);
  float* hf     = (float*)alloc((size_t)Bb * H_ * 4);
  float* cf     = (float*)alloc((size_t)Bb * H_ * 4);
  float* decf   = (float*)alloc((size_t)Bb * H2_ * 4);
  float* ctx    = (float*)alloc((size_t)Bb * H2_ * 4);
  float* cov    = (float*)alloc((size_t)Bb * TK_ * 4);
  float* attnw  = (float*)alloc((size_t)Bb * TK_ * 4);
  float* pgw    = (float*)alloc((size_t)Bb * 4);
  float* out1f  = (float*)alloc((size_t)Bb * H_ * 4);

  // output layout: outputs, h, c, attns, p_gens, coverages (flat, f32)
  float* out     = (float*)d_out;
  float* out_h   = out + (size_t)Bb * TD_ * VX_;
  float* out_c   = out_h + (size_t)Bb * H_;
  float* out_at  = out_c + (size_t)Bb * H_;
  float* out_pg  = out_at + (size_t)Bb * TD_ * TK_;
  float* out_cov = out_pg + (size_t)Bb * TD_;

  auto cvt = [&](const float* s, bf16* d, size_t n) {
    int blocks = (int)((n + 255) / 256);
    if (blocks > 16384) blocks = 16384;
    k_cvt<<<blocks, 256, 0, stream>>>(s, d, (int)n);
  };
  cvt(W2,  W2b,  (size_t)V_ * H_);
  cvt(Wx,  Wxb,  (size_t)E_ * KX_);
  cvt(Wp,  Wpb,  (size_t)H2_ * H2_);
  cvt(W1,  W1b,  (size_t)H_ * KX_);
  cvt(Wenc, Wencb, (size_t)H2_ * H2_);
  cvt(enc, encb, (size_t)Bb * TK_ * H2_);
  k_wcat<<<8192, 256, 0, stream>>>(W_ih, W_hh, Wcatb);
  k_bsum<<<(G4_ + 255) / 256, 256, 0, stream>>>(b_ih, b_hh, bsum);

  // enc_feature = encoder_outputs @ Wenc.T : M=12800, N=1024, K=1024
  k_gemm_bf16<<<dim3(H2_ / 16, (Bb * TK_) / 32), 32, 0, stream>>>(
      encb, Wencb, nullptr, encf, nullptr, 0, /*K=*/H2_, /*ldD=*/H2_);

  k_init<<<(Bb * H2_ + 255) / 256, 256, 0, stream>>>(h0, c0, cov_in,
                                                     hf, cf, ctx, cov, A2);

  for (int step = 0; step < TD_; ++step) {
    k_xcat<<<(Bb * KX_ + 255) / 256, 256, 0, stream>>>(ctx, inputs, emb, xcat, step);
    // x = [ctx|emb] @ Wx.T + bx : M=32,N=512,K=1536 (bf16 mirror -> A2[:, :512])
    k_gemm_bf16<<<dim3(E_ / 16, 1), 32, 0, stream>>>(
        xcat, Wxb, bx, xf, A2, /*ldDbf=*/H2_, /*K=*/KX_, /*ldD=*/E_);
    // gates = [x|h] @ [W_ih|W_hh].T + (b_ih+b_hh) : M=32,N=2048,K=1024
    k_gemm_bf16<<<dim3(G4_ / 16, 1), 32, 0, stream>>>(
        A2, Wcatb, bsum, gates, nullptr, 0, /*K=*/H2_, /*ldD=*/G4_);
    k_lstm<<<(Bb * H_ + 255) / 256, 256, 0, stream>>>(
        gates, hf, cf, A2, Ash, Ahc, out_h, out_c);
    // dec_fea = state_hat @ Wp.T + bp : M=32,N=1024,K=1024
    k_gemm_bf16<<<dim3(H2_ / 16, 1), 32, 0, stream>>>(
        Ash, Wpb, bp, decf, nullptr, 0, /*K=*/H2_, /*ldD=*/H2_);
    k_attn<<<Bb, 256, 0, stream>>>(
        encf, decf, Wc, v_w, enc, mask, hf, cf, xf, Wpg, bpg,
        cov, ctx, Ahc, attnw, out_at, out_cov, pgw, out_pg, step);
    // out1 = [h|c_t] @ W1.T + b1 : M=32,N=512,K=1536 (bf16 mirror out1b)
    k_gemm_bf16<<<dim3(H_ / 16, 1), 32, 0, stream>>>(
        Ahc, W1b, b1, out1f, out1b, /*ldDbf=*/H_, /*K=*/KX_, /*ldD=*/H_);
    // logits = out1 @ W2.T + b2 : M=32,N=50000,K=512  (the dominant GEMM)
    k_gemm_bf16<<<dim3(V_ / 16, 1), 32, 0, stream>>>(
        out1b, W2b, b2, logits, nullptr, 0, /*K=*/H_, /*ldD=*/V_);
    k_vocab<<<Bb, 256, 0, stream>>>(logits, pgw, attnw, ebev, out, step);
  }
}
